// GraphTransformerConv_24876450578533
// MI455X (gfx1250) — matrix operand order repaired and checked by reading, not compiled, not run on hardware
//
#include <hip/hip_runtime.h>
#include <hip/hip_bf16.h>

// ---------------------------------------------------------------------------
// GraphTransformerConv on MI455X (gfx1250).
// All matmuls -> v_wmma_f32_16x16x32_bf16; flash attention (2 q-tiles/wave);
// edge MLP with 64-edge LDS tiles, async global->LDS gather (ASYNCcnt) and
// register-resident LayerNorm; fused output proj + residual + LayerNorm.
// GEMM K-loops use ping-pong double buffering (manual unroll-by-2): loads for
// one fragment set are issued while the other set feeds the WMMAs, with no
// rotation copies.
// ---------------------------------------------------------------------------

typedef __bf16 bf16;
typedef __attribute__((ext_vector_type(16))) bf16  v16bf;
typedef __attribute__((ext_vector_type(8)))  bf16  v8bf;
typedef __attribute__((ext_vector_type(8)))  float v8f;

#define NN   4096
#define CC   256
#define OUTC 256
#define EE   131072
#define HH   8
#define DHH  32
#define ET   64        // edges per block in k_edge

#define WMMA_BF16(a, b, c) \
  __builtin_amdgcn_wmma_f32_16x16x32_bf16(false, (a), false, (b), (short)0, (c), false, false)

// A fragment (16x32, MxK) per CDNA5 ISA layout:
//   lane m=L&15, khalf=L>>4 holds K = khalf*8 + [0..7] and 16 + khalf*8 + [0..7]
__device__ inline v16bf load_a_frag(const bf16* A, int lda, int lane) {
  const int m  = lane & 15;
  const int kh = (lane >> 4) & 1;
  const bf16* p = A + m * lda + kh * 8;
  v8bf lo = *(const v8bf*)(p);
  v8bf hi = *(const v8bf*)(p + 16);
  v16bf f;
#pragma unroll
  for (int i = 0; i < 8; ++i) { f[i] = lo[i]; f[8 + i] = hi[i]; }
  return f;
}

// B fragment for D = A * W^T: logical B[k][n] = W[n][k]; column n=L&15,
// khalf=L>>4 holds K = khalf*8 + [0..7] and 16 + khalf*8 + [0..7].
__device__ inline v16bf load_bt_frag(const bf16* W, int ldw, int lane) {
  const int n  = lane & 15;
  const int kq = (lane >> 4) & 1;
  const bf16* p = W + n * ldw + kq * 8;
  v8bf lo = *(const v8bf*)(p);
  v8bf hi = *(const v8bf*)(p + 16);
  v16bf f;
#pragma unroll
  for (int i = 0; i < 8; ++i) { f[i] = lo[i]; f[8 + i] = hi[i]; }
  return f;
}

// ---------------------------------------------------------------------------
// prep: f32 -> bf16 conversion, zero-fill
// ---------------------------------------------------------------------------
__global__ void k_cvt_bf16(const float* __restrict__ src, bf16* __restrict__ dst, int n) {
  int i = blockIdx.x * blockDim.x + threadIdx.x;
  if (i < n) dst[i] = (bf16)src[i];
}
__global__ void k_zero_f32(float* __restrict__ p, int n) {
  int i = blockIdx.x * blockDim.x + threadIdx.x;
  if (i < n) p[i] = 0.0f;
}

// ---------------------------------------------------------------------------
// QKV projection: qkv = x @ in_proj_w.T + b   ([4096,256] x [768,256]^T)
// Each wave: 1 M-tile x 4 N-tiles; ping-pong pipelined K-loop (8 steps).
// q,k -> [H][N][DH]; v -> transposed [H][DH][N] so P@V is also X@W.T-shaped.
// ---------------------------------------------------------------------------
__global__ __launch_bounds__(128) void k_qkv(const bf16* __restrict__ xb,
                                             const bf16* __restrict__ w_in,
                                             const float* __restrict__ b_in,
                                             bf16* __restrict__ qb,
                                             bf16* __restrict__ kb,
                                             bf16* __restrict__ vtb) {
  const int lane = threadIdx.x & 31;
  const int wid  = blockIdx.x * 4 + (threadIdx.x >> 5);
  const int mt = wid / 12, ng = wid % 12;      // 256 M-tiles x 12 N-groups
  const int m0 = mt * 16, nb = ng * 64;
  const v8f vzero = {};
  v8f acc[4] = {vzero, vzero, vzero, vzero};

  auto loadf = [&](int k0, v16bf& a, v16bf* b) {
    a = load_a_frag(xb + m0 * CC + k0, CC, lane);
#pragma unroll
    for (int t = 0; t < 4; ++t)
      b[t] = load_bt_frag(w_in + (nb + t * 16) * CC + k0, CC, lane);
  };
  auto mma = [&](v16bf a, v16bf* b) {
#pragma unroll
    for (int t = 0; t < 4; ++t) acc[t] = WMMA_BF16(a, b[t], acc[t]);
  };

  v16bf aA, aB, bA[4], bB[4];
  loadf(0, aA, bA);
  for (int k0 = 0; k0 < CC - 64; k0 += 64) {
    loadf(k0 + 32, aB, bB);
    mma(aA, bA);
    loadf(k0 + 64, aA, bA);
    mma(aB, bB);
  }
  loadf(CC - 32, aB, bB);
  mma(aA, bA);
  mma(aB, bB);

  const int half = lane >> 4;
#pragma unroll
  for (int t = 0; t < 4; ++t) {
    const int n = nb + t * 16 + (lane & 15);
    const float bias = b_in[n];
#pragma unroll
    for (int r = 0; r < 8; ++r) {
      const int row = m0 + r + 8 * half;       // C/D layout: M = r + 8*(lane>=16)
      const float v = acc[t][r] + bias;
      if (n < CC) {
        qb[(((n >> 5) * NN) + row) * DHH + (n & 31)] = (bf16)v;
      } else if (n < 2 * CC) {
        const int n2 = n - CC;
        kb[(((n2 >> 5) * NN) + row) * DHH + (n2 & 31)] = (bf16)v;
      } else {
        const int n2 = n - 2 * CC;
        vtb[(((n2 >> 5) * DHH) + (n2 & 31)) * (size_t)NN + row] = (bf16)v;
      }
    }
  }
}

// ---------------------------------------------------------------------------
// Flash attention: each wave owns TWO 16-row q tiles of one head and streams
// all 4096 keys in 32-chunks; ping-pong pipelined K/V fragments so the next
// chunk's loads hide under the softmax VALU block.
// Online softmax with 16-lane shfl reductions; P restaged via 1KB LDS patch.
// ---------------------------------------------------------------------------
__global__ __launch_bounds__(128) void k_attn(const bf16* __restrict__ qb,
                                              const bf16* __restrict__ kb,
                                              const bf16* __restrict__ vtb,
                                              bf16* __restrict__ ctx) {
  __shared__ bf16 p_lds[4][2][16][32];
  const int lane = threadIdx.x & 31;
  const int wiv  = threadIdx.x >> 5;
  const int wid  = blockIdx.x * 4 + wiv;
  const int h  = wid >> 7;                     // 128 q-pairs per head
  const int qt = wid & 127;
  const int m0 = qt * 32;
  const bf16* qh = qb  + (size_t)h * NN * DHH;
  const bf16* kh = kb  + (size_t)h * NN * DHH;
  const bf16* vh = vtb + (size_t)h * DHH * NN;

  v16bf qf[2];
  qf[0] = load_a_frag(qh + (m0 +  0) * DHH, DHH, lane);
  qf[1] = load_a_frag(qh + (m0 + 16) * DHH, DHH, lane);
  const v8f vzero = {};
  float rm[2][8], rl[2][8];
  v8f acc[2][2];
#pragma unroll
  for (int i = 0; i < 2; ++i) {
    acc[i][0] = vzero; acc[i][1] = vzero;
#pragma unroll
    for (int r = 0; r < 8; ++r) { rm[i][r] = -3.0e38f; rl[i][r] = 0.0f; }
  }
  const float scale = 0.17677669529663687f;    // 1/sqrt(DH)
  const int half = lane >> 4;
  const int nloc = lane & 15;

  auto loadkv = [&](int kc, v16bf& k0f, v16bf& k1f, v16bf& v0f, v16bf& v1f) {
    k0f = load_bt_frag(kh + (kc +  0) * DHH, DHH, lane);
    k1f = load_bt_frag(kh + (kc + 16) * DHH, DHH, lane);
    v0f = load_bt_frag(vh + (size_t) 0 * NN + kc, NN, lane);
    v1f = load_bt_frag(vh + (size_t)16 * NN + kc, NN, lane);
  };
  auto process = [&](v16bf bk0, v16bf bk1, v16bf bv0, v16bf bv1) {
#pragma unroll
    for (int i = 0; i < 2; ++i) {
      v8f s0 = WMMA_BF16(qf[i], bk0, vzero);
      v8f s1 = WMMA_BF16(qf[i], bk1, vzero);
      bf16 (*pt)[32] = p_lds[wiv][i];
#pragma unroll
      for (int r = 0; r < 8; ++r) {
        float a = s0[r] * scale, b = s1[r] * scale;
        float mx = fmaxf(a, b);
#pragma unroll
        for (int off = 8; off >= 1; off >>= 1) mx = fmaxf(mx, __shfl_xor(mx, off, 32));
        const float mnew = fmaxf(rm[i][r], mx);
        const float corr = __expf(rm[i][r] - mnew);
        const float p0 = __expf(a - mnew);
        const float p1 = __expf(b - mnew);
        float ps = p0 + p1;
#pragma unroll
        for (int off = 8; off >= 1; off >>= 1) ps += __shfl_xor(ps, off, 32);
        rl[i][r] = rl[i][r] * corr + ps;
        rm[i][r] = mnew;
        acc[i][0][r] *= corr;
        acc[i][1][r] *= corr;
        const int m = r + 8 * half;
        pt[m][nloc]      = (bf16)p0;
        pt[m][16 + nloc] = (bf16)p1;
      }
      v16bf pf = load_a_frag(&pt[0][0], 32, lane);
      acc[i][0] = WMMA_BF16(pf, bv0, acc[i][0]);
      acc[i][1] = WMMA_BF16(pf, bv1, acc[i][1]);
    }
  };

  v16bf kA0, kA1, vA0, vA1, kB0, kB1, vB0, vB1;
  loadkv(0, kA0, kA1, vA0, vA1);
  for (int kb0 = 0; kb0 < NN - 64; kb0 += 64) {
    loadkv(kb0 + 32, kB0, kB1, vB0, vB1);
    process(kA0, kA1, vA0, vA1);
    loadkv(kb0 + 64, kA0, kA1, vA0, vA1);
    process(kB0, kB1, vB0, vB1);
  }
  loadkv(NN - 32, kB0, kB1, vB0, vB1);
  process(kA0, kA1, vA0, vA1);
  process(kB0, kB1, vB0, vB1);

#pragma unroll
  for (int i = 0; i < 2; ++i)
#pragma unroll
    for (int r = 0; r < 8; ++r) {
      const int row = m0 + i * 16 + r + 8 * half;
      const float inv = 1.0f / rl[i][r];
      ctx[row * CC + h * DHH + nloc]      = (bf16)(acc[i][0][r] * inv);
      ctx[row * CC + h * DHH + 16 + nloc] = (bf16)(acc[i][1][r] * inv);
    }
}

// ---------------------------------------------------------------------------
// Attention output projection: x_attn = ctx @ attn_out_w.T + b (f32 out).
// Each wave: 1 M-tile x 4 N-tiles; ping-pong pipelined K-loop.
// ---------------------------------------------------------------------------
__global__ __launch_bounds__(128) void k_attn_proj(const bf16* __restrict__ ctx,
                                                   const bf16* __restrict__ w_ao,
                                                   const float* __restrict__ b_ao,
                                                   float* __restrict__ x_attn) {
  const int lane = threadIdx.x & 31;
  const int wid  = blockIdx.x * 4 + (threadIdx.x >> 5);
  const int mt = wid >> 2, ng = wid & 3;
  const int m0 = mt * 16, nb = ng * 64;
  const v8f vzero = {};
  v8f acc[4] = {vzero, vzero, vzero, vzero};

  auto loadf = [&](int k0, v16bf& a, v16bf* b) {
    a = load_a_frag(ctx + m0 * CC + k0, CC, lane);
#pragma unroll
    for (int t = 0; t < 4; ++t)
      b[t] = load_bt_frag(w_ao + (nb + t * 16) * CC + k0, CC, lane);
  };
  auto mma = [&](v16bf a, v16bf* b) {
#pragma unroll
    for (int t = 0; t < 4; ++t) acc[t] = WMMA_BF16(a, b[t], acc[t]);
  };

  v16bf aA, aB, bA[4], bB[4];
  loadf(0, aA, bA);
  for (int k0 = 0; k0 < CC - 64; k0 += 64) {
    loadf(k0 + 32, aB, bB);
    mma(aA, bA);
    loadf(k0 + 64, aA, bA);
    mma(aB, bB);
  }
  loadf(CC - 32, aB, bB);
  mma(aA, bA);
  mma(aB, bB);

  const int half = lane >> 4;
#pragma unroll
  for (int t = 0; t < 4; ++t) {
    const int n = nb + t * 16 + (lane & 15);
    const float bias = b_ao[n];
#pragma unroll
    for (int r = 0; r < 8; ++r)
      x_attn[(m0 + r + 8 * half) * CC + n] = acc[t][r] + bias;
  }
}

// ---------------------------------------------------------------------------
// Edge MLP, fused per 64-edge tile (512 threads = 16 waves):
//   x[row]/x[col] slices gathered bf16->bf16 with GLOBAL_LOAD_ASYNC_TO_LDS_B128
//   (ASYNCcnt), edge_attr converted f32->bf16 via VALU;
//   [64,768]x[768,256]: wave w -> M-tile (w&3), 4 N-tiles at (w>>2)*64,
//   ping-pong pipelined K-loop (A from LDS, B from L2-resident weights);
//   LayerNorm stats from registers (shfl + ds_add_f32), ReLU, atomic scatter.
// Weight L2 traffic: 2048 blocks x 393KB ~= 0.8GB.
// ---------------------------------------------------------------------------
__global__ __launch_bounds__(512) void k_edge(const bf16*  __restrict__ xb,
                                              const int*   __restrict__ eidx,
                                              const float* __restrict__ eattr,
                                              const bf16*  __restrict__ w_e,
                                              const float* __restrict__ b_e,
                                              const float* __restrict__ g1,
                                              const float* __restrict__ bb1,
                                              float* __restrict__ node_msg) {
  __shared__ bf16  a_tile[ET][3 * CC];         // 96 KB, LDS offset 0
  __shared__ float s_sum[ET];
  __shared__ float s_ss[ET];
  const int tid = threadIdx.x;
  const int e0  = blockIdx.x * ET;

  // --- async gather of the two x slices (16B per lane, straight to LDS) ---
  for (int t4 = tid; t4 < ET * 64; t4 += 512) {
    const int row   = t4 >> 6;
    const int q     = t4 & 63;
    const int which = q >> 5;                  // 0: x[row], 1: x[col]
    const int chunk = q & 31;                  // 16B chunk of the 512B row
    const int node  = eidx[(which ? EE : 0) + e0 + row];
    const bf16* src = xb + (size_t)node * CC + chunk * 8;
    const unsigned dst = (unsigned)(row * (3 * CC * 2) + which * 512 + chunk * 16);
    asm volatile("global_load_async_to_lds_b128 %0, %1, off"
                 :: "v"(dst), "v"((unsigned long long)(uintptr_t)src)
                 : "memory");
  }
  // --- edge_attr slice: f32 -> bf16 conversion ---
  for (int t = tid; t < ET * CC; t += 512) {
    const int row = t >> 8;
    const int c   = t & 255;
    a_tile[row][2 * CC + c] = (bf16)eattr[(size_t)(e0 + row) * CC + c];
  }
  if (tid < ET) { s_sum[tid] = 0.0f; s_ss[tid] = 0.0f; }
  asm volatile("s_wait_asynccnt 0" ::: "memory");
  __syncthreads();

  // --- GEMM: wave w -> rows mt..mt+15, cols nb..nb+63, ping-pong K-loop ---
  const int lane = tid & 31;
  const int w    = tid >> 5;
  const int mt   = (w & 3) * 16;
  const int nb   = (w >> 2) * 64;
  const int half = lane >> 4;
  const v8f vzero = {};
  v8f acc[4] = {vzero, vzero, vzero, vzero};

  auto loadf = [&](int k0, v16bf& a, v16bf* b) {
    a = load_a_frag(&a_tile[mt][0] + k0, 3 * CC, lane);
#pragma unroll
    for (int t = 0; t < 4; ++t)
      b[t] = load_bt_frag(w_e + (nb + t * 16) * (3 * CC) + k0, 3 * CC, lane);
  };
  auto mma = [&](v16bf a, v16bf* b) {
#pragma unroll
    for (int t = 0; t < 4; ++t) acc[t] = WMMA_BF16(a, b[t], acc[t]);
  };

  v16bf aA, aB, bA[4], bB[4];
  loadf(0, aA, bA);
  for (int k0 = 0; k0 < 3 * CC - 64; k0 += 64) {
    loadf(k0 + 32, aB, bB);
    mma(aA, bA);
    loadf(k0 + 64, aA, bA);
    mma(aB, bB);
  }
  loadf(3 * CC - 32, aB, bB);
  mma(aA, bA);
  mma(aB, bB);

#pragma unroll
  for (int t = 0; t < 4; ++t) {
    const float bias = b_e[nb + t * 16 + (lane & 15)];
#pragma unroll
    for (int r = 0; r < 8; ++r) acc[t][r] += bias;
  }

  // --- per-row LayerNorm stats: in-register partials -> shfl -> LDS atomics ---
#pragma unroll
  for (int r = 0; r < 8; ++r) {
    float ps = 0.0f, pss = 0.0f;
#pragma unroll
    for (int t = 0; t < 4; ++t) { const float v = acc[t][r]; ps += v; pss += v * v; }
#pragma unroll
    for (int off = 8; off >= 1; off >>= 1) {
      ps  += __shfl_xor(ps,  off, 32);
      pss += __shfl_xor(pss, off, 32);
    }
    if ((lane & 15) == 0) {
      const int row = mt + r + 8 * half;
      atomicAdd(&s_sum[row], ps);
      atomicAdd(&s_ss[row],  pss);
    }
  }
  __syncthreads();

  // --- normalize + ReLU + scatter-add (global f32 atomics) ---
#pragma unroll
  for (int r = 0; r < 8; ++r) {
    const int row  = mt + r + 8 * half;
    const float mean = s_sum[row] * (1.0f / OUTC);
    const float var  = s_ss[row] * (1.0f / OUTC) - mean * mean;
    const float rstd = rsqrtf(var + 1e-5f);
    const int   dest = eidx[EE + e0 + row];    // segment_sum over col index
#pragma unroll
    for (int t = 0; t < 4; ++t) {
      const int n = nb + t * 16 + (lane & 15);
      float v = (acc[t][r] - mean) * rstd * g1[n] + bb1[n];
      v = fmaxf(v, 0.0f);
      atomicAdd(&node_msg[(size_t)dest * OUTC + n], v);
    }
  }
}

// ---------------------------------------------------------------------------
// Output projection + residual + final LayerNorm, fused per 16-node tile.
// ---------------------------------------------------------------------------
__global__ __launch_bounds__(256) void k_out(const float* __restrict__ x_attn,
                                             const float* __restrict__ node_msg,
                                             const float* __restrict__ x,
                                             const bf16*  __restrict__ w_o,
                                             const float* __restrict__ b_o,
                                             const float* __restrict__ g2,
                                             const float* __restrict__ bb2,
                                             float* __restrict__ out) {
  __shared__ bf16  a_tile[16][2 * CC];         // 16 KB
  __shared__ float o_tile[16][OUTC];           // 16 KB
  const int tid = threadIdx.x;
  const int m0  = blockIdx.x * 16;

  for (int idx = tid; idx < 16 * 2 * CC; idx += 256) {
    const int rr = idx / (2 * CC);
    const int c  = idx % (2 * CC);
    const float v = (c < CC) ? x_attn[(m0 + rr) * CC + c]
                             : node_msg[(m0 + rr) * OUTC + (c - CC)];
    a_tile[rr][c] = (bf16)v;
  }
  __syncthreads();

  const int lane = tid & 31;
  const int w    = tid >> 5;
  const int half = lane >> 4;
  const v8f vzero = {};
  v8f acc[2] = {vzero, vzero};

  auto loadf = [&](int k0, v16bf& a, v16bf* b) {
    a = load_a_frag(&a_tile[0][0] + k0, 2 * CC, lane);
#pragma unroll
    for (int t = 0; t < 2; ++t)
      b[t] = load_bt_frag(w_o + ((2 * w + t) * 16) * (2 * CC) + k0, 2 * CC, lane);
  };
  auto mma = [&](v16bf a, v16bf* b) {
#pragma unroll
    for (int t = 0; t < 2; ++t) acc[t] = WMMA_BF16(a, b[t], acc[t]);
  };

  v16bf aA, aB, bA[2], bB[2];
  loadf(0, aA, bA);
  for (int k0 = 0; k0 < 2 * CC - 64; k0 += 64) {
    loadf(k0 + 32, aB, bB);
    mma(aA, bA);
    loadf(k0 + 64, aA, bA);
    mma(aB, bB);
  }
  loadf(2 * CC - 32, aB, bB);
  mma(aA, bA);
  mma(aB, bB);

#pragma unroll
  for (int t = 0; t < 2; ++t) {
    const int n = (2 * w + t) * 16 + (lane & 15);
    const float bias = b_o[n];
#pragma unroll
    for (int r = 0; r < 8; ++r) {
      const int row = r + 8 * half;
      o_tile[row][n] = acc[t][r] + bias + x[(m0 + row) * CC + n];  // residual
    }
  }
  __syncthreads();

  const int row = tid >> 4;
  const int sub = tid & 15;
  float s = 0.f, ss = 0.f;
#pragma unroll
  for (int j = 0; j < 16; ++j) {
    const float v = o_tile[row][sub + 16 * j];
    s += v; ss += v * v;
  }
#pragma unroll
  for (int off = 8; off >= 1; off >>= 1) {
    s  += __shfl_xor(s,  off, 32);
    ss += __shfl_xor(ss, off, 32);
  }
  const float mean = s * (1.0f / OUTC);
  const float var  = ss * (1.0f / OUTC) - mean * mean;
  const float rstd = rsqrtf(var + 1e-5f);
#pragma unroll
  for (int j = 0; j < 16; ++j) {
    const int c = sub + 16 * j;
    out[(m0 + row) * OUTC + c] = (o_tile[row][c] - mean) * rstd * g2[c] + bb2[c];
  }
}

// ---------------------------------------------------------------------------
// Workspace layout (bytes) — ~19.1 MB total, 256B aligned
// ---------------------------------------------------------------------------
#define WS_XBF    (size_t)0          // 4096*256 bf16          = 2097152
#define WS_WIN    (size_t)2097152    // 768*256 bf16           =  393216
#define WS_WAO    (size_t)2490368    // 256*256 bf16           =  131072
#define WS_WE     (size_t)2621440    // 256*768 bf16           =  393216
#define WS_WO     (size_t)3014656    // 256*512 bf16           =  262144
#define WS_Q      (size_t)3276800    // [H][N][DH] bf16        = 2097152
#define WS_K      (size_t)5373952
#define WS_VT     (size_t)7471104    // [H][DH][N] bf16
#define WS_CTX    (size_t)9568256    // [N][C] bf16
#define WS_XATT   (size_t)11665408   // [N][C] f32             = 4194304
#define WS_NMSG   (size_t)15859712   // [N][OUT] f32           = 4194304

extern "C" void kernel_launch(void* const* d_in, const int* in_sizes, int n_in,
                              void* d_out, int out_size, void* d_ws, size_t ws_size,
                              hipStream_t stream) {
  const float* x     = (const float*)d_in[0];
  const int*   eidx  = (const int*)  d_in[1];
  const float* eattr = (const float*)d_in[2];
  const float* in_w  = (const float*)d_in[3];
  const float* in_b  = (const float*)d_in[4];
  const float* ao_w  = (const float*)d_in[5];
  const float* ao_b  = (const float*)d_in[6];
  const float* e_w   = (const float*)d_in[7];
  const float* e_b   = (const float*)d_in[8];
  const float* ln1_g = (const float*)d_in[9];
  const float* ln1_b = (const float*)d_in[10];
  const float* o_w   = (const float*)d_in[11];
  const float* o_b   = (const float*)d_in[12];
  const float* ln2_g = (const float*)d_in[13];
  const float* ln2_b = (const float*)d_in[14];

  char* ws = (char*)d_ws;
  bf16*  xb   = (bf16*)(ws + WS_XBF);
  bf16*  winb = (bf16*)(ws + WS_WIN);
  bf16*  waob = (bf16*)(ws + WS_WAO);
  bf16*  web  = (bf16*)(ws + WS_WE);
  bf16*  wob  = (bf16*)(ws + WS_WO);
  bf16*  qb   = (bf16*)(ws + WS_Q);
  bf16*  kbf  = (bf16*)(ws + WS_K);
  bf16*  vtb  = (bf16*)(ws + WS_VT);
  bf16*  ctx  = (bf16*)(ws + WS_CTX);
  float* xatt = (float*)(ws + WS_XATT);
  float* nmsg = (float*)(ws + WS_NMSG);

  // --- prep: bf16 copies, zero the message accumulator
  k_cvt_bf16<<<(NN * CC + 255) / 256, 256, 0, stream>>>(x,    xb,   NN * CC);
  k_cvt_bf16<<<(3 * CC * CC + 255) / 256, 256, 0, stream>>>(in_w, winb, 3 * CC * CC);
  k_cvt_bf16<<<(CC * CC + 255) / 256, 256, 0, stream>>>(ao_w, waob, CC * CC);
  k_cvt_bf16<<<(OUTC * 3 * CC + 255) / 256, 256, 0, stream>>>(e_w, web, OUTC * 3 * CC);
  k_cvt_bf16<<<(OUTC * 2 * CC + 255) / 256, 256, 0, stream>>>(o_w, wob, OUTC * 2 * CC);
  k_zero_f32<<<(NN * OUTC + 255) / 256, 256, 0, stream>>>(nmsg, NN * OUTC);

  // --- dense self-attention path
  k_qkv<<<(256 * 12) / 4, 128, 0, stream>>>(xb, winb, in_b, qb, kbf, vtb);
  k_attn<<<(HH * 128) / 4, 128, 0, stream>>>(qb, kbf, vtb, ctx);
  k_attn_proj<<<(256 * 4) / 4, 128, 0, stream>>>(ctx, waob, ao_b, xatt);

  // --- edge message path
  k_edge<<<EE / ET, 512, 0, stream>>>(xb, eidx, eattr, web, e_b, ln1_g, ln1_b, nmsg);

  // --- combine + project + residual + LN
  k_out<<<NN / 16, 256, 0, stream>>>(xatt, nmsg, x, wob, o_b, ln2_g, ln2_b, (float*)d_out);
}
